// MultiMaskCrossBlock2_22960895165090
// MI455X (gfx1250) — matrix operand order repaired and checked by reading, not compiled, not run on hardware
//
#include <hip/hip_runtime.h>

// ---------------------------------------------------------------------------
// MI455X / gfx1250 implementation of MultiMaskCrossBlock.
// All channel-mixing GEMMs run on v_wmma_f32_16x16x32_f16 (f16 in, f32 acc).
// LDS tiles are laid out so every WMMA fragment is 2x contiguous, 16B-aligned
// ds_load_b128 per lane (no strided u16 gathers).
// ---------------------------------------------------------------------------

typedef __attribute__((ext_vector_type(16))) _Float16 v16h;
typedef __attribute__((ext_vector_type(8)))  float    v8f;

#define FLAG_GELU 1
#define FLAG_RES  2
#define FLAG_LN   4
#define FLAG_BIAS 8

static constexpr int Bn  = 2;
static constexpr int Cch = 64;
static constexpr int Sd  = 32;
static constexpr int Ls  = Sd * Sd * Sd;   // 32768
static constexpr int KCc = 256;            // 4 modalities * 64
static constexpr int LCHUNK = 1024;        // split-K chunk for q*k^T

__device__ __forceinline__ float gelu_exact(float x) {
    return 0.5f * x * (1.0f + erff(x * 0.70710678118654752f));
}

// ---------------------------------------------------------------------------
// Pointwise GEMM: Y[b,o,n] = act(LN(sum_i W[o,i]*X[b,i,n] + bias) (+ res))
// One workgroup: all Co rows x 64 spatial columns. Wave w owns M-tile w.
// Ws row-major [Co][Ci]; Xs column-major [64][Ci+16] so both A and B
// fragments are two contiguous 8-half runs per lane.
// ---------------------------------------------------------------------------
__global__ void gemm_pw_kernel(const float* __restrict__ X, const float* __restrict__ W,
                               const float* __restrict__ bias, const float* __restrict__ gamma,
                               const float* __restrict__ beta, const float* __restrict__ res,
                               float* __restrict__ Y,
                               int Ci, int Co, int y_ctot, int y_coff,
                               long w_bstride, int flags)
{
    extern __shared__ char smem[];
    const int XPad = Ci + 16;                  // halves; keeps 16B alignment, spreads banks
    _Float16* Ws = (_Float16*)smem;            // [Co][Ci] f16, row-major
    _Float16* Xs = Ws + Co * Ci;               // [64][XPad] f16, column-major
    float*    Ys = (float*)(Xs + 64 * XPad);   // [Co][64] f32
    float*    mu = Ys + Co * 64;               // [64]
    float*    rs = mu + 64;                    // [64]

    const int tid  = threadIdx.x;
    const int nthr = blockDim.x;
    const int b    = blockIdx.y;
    const int n0   = blockIdx.x * 64;

    // Stage weights (optionally per-batch for attn*v) and a 64-column X tile.
    const float* Wb = W + (long)b * w_bstride;
    for (int idx = tid; idx < Co * Ci; idx += nthr)
        Ws[idx] = (_Float16)Wb[idx];
    for (int idx = tid; idx < Ci * 64; idx += nthr) {
        int i = idx >> 6, n = idx & 63;        // coalesced global read over n
        Xs[n * XPad + i] = (_Float16)X[((long)(b * Ci + i)) * Ls + n0 + n];
    }
    __syncthreads();

    const int w    = tid >> 5;     // wave id == M tile
    const int lane = tid & 31;
    const int h    = lane >> 4;    // half-wave select
    const int ml   = lane & 15;
    const int rowA = w * 16 + ml;  // A-matrix row (both halves hold same row)

    v8f acc[4];
#pragma unroll
    for (int nt = 0; nt < 4; ++nt) acc[nt] = (v8f){0.f,0.f,0.f,0.f,0.f,0.f,0.f,0.f};

    for (int kt = 0; kt < Ci; kt += 32) {
        v16h a;
#pragma unroll
        for (int j = 0; j < 16; ++j) {         // A 16x32 f16 layout: K = 8h + (j<8 ? j : j+8)
            int kk = kt + 8 * h + (j < 8 ? j : j + 8);
            a[j] = Ws[rowA * Ci + kk];         // two contiguous 8-half runs -> 2x b128
        }
#pragma unroll
        for (int nt = 0; nt < 4; ++nt) {
            const int col = nt * 16 + ml;
            v16h bf;
#pragma unroll
            for (int j = 0; j < 16; ++j)       // B 32x16 f16 layout: K = 8h + (j<8 ? j : j+8)
                bf[j] = Xs[col * XPad + kt + 8 * h + (j < 8 ? j : j + 8)];
            acc[nt] = __builtin_amdgcn_wmma_f32_16x16x32_f16(false, a, false, bf,
                                                             (short)0, acc[nt], false, false);
        }
    }
#pragma unroll
    for (int nt = 0; nt < 4; ++nt) {
#pragma unroll
        for (int r = 0; r < 8; ++r) {          // C/D layout: M = r + 8h, N = ml
            int M = w * 16 + h * 8 + r;
            float val = acc[nt][r];
            if (flags & FLAG_BIAS) val += bias[M];
            Ys[M * 64 + nt * 16 + ml] = val;
        }
    }
    __syncthreads();

    if (flags & FLAG_LN) {
        if (tid < 64) {                        // LN over channels, per column
            float s = 0.f;
            for (int m = 0; m < Co; ++m) s += Ys[m * 64 + tid];
            float mean = s / (float)Co;
            float v = 0.f;
            for (int m = 0; m < Co; ++m) { float d = Ys[m * 64 + tid] - mean; v += d * d; }
            mu[tid] = mean;
            rs[tid] = rsqrtf(v / (float)Co + 1e-6f);
        }
        __syncthreads();
    }

    for (int idx = tid; idx < Co * 64; idx += nthr) {
        int o = idx >> 6, n = idx & 63;
        float y = Ys[idx];
        if (flags & FLAG_LN)   y = gamma[o] * (y - mu[n]) * rs[n] + beta[o];
        if (flags & FLAG_RES)  y += res[((long)(b * Co + o)) * Ls + n0 + n];
        if (flags & FLAG_GELU) y = gelu_exact(y);
        Y[((long)(b * y_ctot + y_coff + o)) * Ls + n0 + n] = y;
    }
}

// ---------------------------------------------------------------------------
// Grouped / depthwise 3x3x3 conv (pad 1) + LN over channels + GELU.
// One workgroup: 64 consecutive spatial positions x all Cout channels.
// ---------------------------------------------------------------------------
__global__ void conv3_ln_gelu_kernel(const float* __restrict__ X, const float* __restrict__ Wc,
                                     const float* __restrict__ bias, const float* __restrict__ gamma,
                                     const float* __restrict__ beta, float* __restrict__ Y,
                                     int Cin, int Cout, int cin_pg, int cout_pg)
{
    extern __shared__ char smem[];
    float* Ys = (float*)smem;                  // [Cout][64]
    float* mu = Ys + Cout * 64;
    float* rs = mu + 64;

    const int tid = threadIdx.x;
    const int b   = blockIdx.y;
    const int l0  = blockIdx.x * 64;
    const int d   = l0 >> 10;                  // 32*32 positions per depth slab
    const int h0  = (l0 >> 5) & 31;
    const int p   = tid & 63;
    const int hh  = h0 + (p >> 5);
    const int ww  = p & 31;

    for (int c = tid >> 6; c < Cout; c += 4) {
        int g = c / cout_pg;
        float acc = bias[c];
        for (int i = 0; i < cin_pg; ++i) {
            const float* xc = X + ((long)(b * Cin + g * cin_pg + i)) * Ls;
            const float* wp = Wc + ((long)(c * cin_pg + i)) * 27;
            int kidx = 0;
            for (int kd = -1; kd <= 1; ++kd) {
                int zd = d + kd; bool okd = (unsigned)zd < 32u;
                for (int kh = -1; kh <= 1; ++kh) {
                    int zh = hh + kh; bool okh = (unsigned)zh < 32u;
                    for (int kw = -1; kw <= 1; ++kw, ++kidx) {
                        int zw = ww + kw;
                        if (okd && okh && (unsigned)zw < 32u)
                            acc += xc[(zd * 32 + zh) * 32 + zw] * wp[kidx];
                    }
                }
            }
        }
        Ys[c * 64 + p] = acc;
    }
    __syncthreads();

    if (tid < 64) {
        float s = 0.f;
        for (int m = 0; m < Cout; ++m) s += Ys[m * 64 + tid];
        float mean = s / (float)Cout;
        float v = 0.f;
        for (int m = 0; m < Cout; ++m) { float dd = Ys[m * 64 + tid] - mean; v += dd * dd; }
        mu[tid] = mean;
        rs[tid] = rsqrtf(v / (float)Cout + 1e-6f);
    }
    __syncthreads();

    for (int idx = tid; idx < Cout * 64; idx += 256) {
        int o = idx >> 6, n = idx & 63;
        float y = Ys[idx];
        y = gamma[o] * (y - mu[n]) * rs[n] + beta[o];
        y = gelu_exact(y);
        Y[((long)(b * Cout + o)) * Ls + l0 + n] = y;
    }
}

// ---------------------------------------------------------------------------
// attn_raw[b,q,kc] += sum_{l in chunk} q[b,q,l] * k[b,kc,l]   (split-K WMMA)
// Block = 8 waves: wave (mt = w&3, nhalf = w>>2) owns 16x128 of the 64x256 out.
// qs/ks stored row-major with K contiguous -> fragments are b128 loads.
// ---------------------------------------------------------------------------
__global__ void qk_kernel(const float* __restrict__ q, const float* __restrict__ k,
                          float* __restrict__ attn)
{
    __shared__ _Float16 qs[64 * 32];
    __shared__ _Float16 ks[256 * 32];

    const int tid  = threadIdx.x;
    const int b    = blockIdx.y;
    const long l_start = (long)blockIdx.x * LCHUNK;
    const int w    = tid >> 5, lane = tid & 31;
    const int h    = lane >> 4, ml = lane & 15;
    const int mt   = w & 3;        // M tile (q channels)
    const int nh   = w >> 2;       // N half (0/1) of 16 N tiles

    v8f acc[8];
#pragma unroll
    for (int i = 0; i < 8; ++i) acc[i] = (v8f){0.f,0.f,0.f,0.f,0.f,0.f,0.f,0.f};

    for (long l = l_start; l < l_start + LCHUNK; l += 32) {
        for (int idx = tid; idx < 64 * 32; idx += 256) {
            int r = idx >> 5, kk = idx & 31;
            qs[idx] = (_Float16)q[((long)(b * 64 + r)) * Ls + l + kk];
        }
        for (int idx = tid; idx < 256 * 32; idx += 256) {
            int r = idx >> 5, kk = idx & 31;
            ks[idx] = (_Float16)k[((long)(b * 256 + r)) * Ls + l + kk];
        }
        __syncthreads();

        v16h a;
#pragma unroll
        for (int j = 0; j < 16; ++j) {
            int kk = 8 * h + (j < 8 ? j : j + 8);
            a[j] = qs[(mt * 16 + ml) * 32 + kk];
        }
#pragma unroll
        for (int nt = 0; nt < 8; ++nt) {
            int col = (nh * 8 + nt) * 16 + ml;     // kc channel
            v16h bf;
#pragma unroll
            for (int j = 0; j < 16; ++j) bf[j] = ks[col * 32 + 8 * h + (j < 8 ? j : j + 8)];
            acc[nt] = __builtin_amdgcn_wmma_f32_16x16x32_f16(false, a, false, bf,
                                                             (short)0, acc[nt], false, false);
        }
        __syncthreads();
    }

    for (int nt = 0; nt < 8; ++nt) {
#pragma unroll
        for (int r = 0; r < 8; ++r) {
            int M = mt * 16 + h * 8 + r;
            int N = (nh * 8 + nt) * 16 + ml;
            atomicAdd(&attn[((long)(b * 64 + M)) * 256 + N], acc[nt][r]);
        }
    }
}

// ---------------------------------------------------------------------------
// scale by L^-1/2, modality mask -> -inf, softmax over kc. One row per wave32.
// ---------------------------------------------------------------------------
__global__ void softmax_mask_kernel(float* __restrict__ attn, const int* __restrict__ mask)
{
    const int row = blockIdx.x * 8 + (threadIdx.x >> 5);
    if (row >= Bn * Cch) return;
    const int b    = row >> 6;
    const int lane = threadIdx.x & 31;
    const float scale = rsqrtf((float)Ls);
    float* rowp = attn + (long)row * 256;

    float v[8]; float mx = -INFINITY;
#pragma unroll
    for (int j = 0; j < 8; ++j) {
        int kc = lane + 32 * j;
        float t = rowp[kc] * scale;
        if (mask[b * 4 + (kc >> 6)] == 0) t = -INFINITY;
        v[j] = t; mx = fmaxf(mx, t);
    }
    for (int off = 16; off >= 1; off >>= 1) mx = fmaxf(mx, __shfl_xor(mx, off, 32));
    float s = 0.f;
#pragma unroll
    for (int j = 0; j < 8; ++j) { v[j] = expf(v[j] - mx); s += v[j]; }
    for (int off = 16; off >= 1; off >>= 1) s += __shfl_xor(s, off, 32);
    float inv = 1.f / s;
#pragma unroll
    for (int j = 0; j < 8; ++j) rowp[lane + 32 * j] = v[j] * inv;
}

// ---------------------------------------------------------------------------
// Host side
// ---------------------------------------------------------------------------
struct DWP { const float *w1,*b1,*g1,*e1,*w2,*b2,*g2,*e2,*w3,*b3,*g3,*e3; };

static DWP getdw(void* const* din, int base) {
    DWP p;
    const float** f = (const float**)&p;
    for (int i = 0; i < 12; ++i) f[i] = (const float*)din[base + i];
    return p;
}

static void launch_gemm(hipStream_t s, const float* X, const float* W, const float* bias,
                        const float* gamma, const float* beta, const float* res, float* Y,
                        int Ci, int Co, int y_ctot, int y_coff, long wbs, int flags)
{
    dim3 grid(Ls / 64, Bn);
    int thr = (Co / 16) * 32;
    size_t lds = 2 * (size_t)(Co * Ci + 64 * (Ci + 16)) + 4 * (size_t)(Co * 64 + 128);
    (void)hipFuncSetAttribute(reinterpret_cast<const void*>(gemm_pw_kernel),
                              hipFuncAttributeMaxDynamicSharedMemorySize, (int)lds);
    gemm_pw_kernel<<<grid, thr, lds, s>>>(X, W, bias, gamma, beta, res, Y,
                                          Ci, Co, y_ctot, y_coff, wbs, flags);
}

static void launch_conv(hipStream_t s, const float* X, const float* Wc, const float* bias,
                        const float* gamma, const float* beta, float* Y,
                        int Cin, int Cout, int cin_pg, int cout_pg)
{
    dim3 grid(Ls / 64, Bn);
    size_t lds = 4 * (size_t)(Cout * 64 + 128);
    (void)hipFuncSetAttribute(reinterpret_cast<const void*>(conv3_ln_gelu_kernel),
                              hipFuncAttributeMaxDynamicSharedMemorySize, (int)lds);
    conv3_ln_gelu_kernel<<<grid, 256, lds, s>>>(X, Wc, bias, gamma, beta, Y,
                                                Cin, Cout, cin_pg, cout_pg);
}

// pw1 -> LN -> GELU -> dw3 -> LN -> GELU -> pw3 -> LN (+res)
// in -> tA -> tB -> out   (tB may alias in; tA must not alias in)
static void dwblock(hipStream_t s, const float* in, const DWP& p, float* tA, float* tB,
                    float* out, int y_ctot, int y_coff, const float* res)
{
    launch_gemm(s, in, p.w1, p.b1, p.g1, p.e1, nullptr, tA, Cch, Cch, Cch, 0, 0,
                FLAG_LN | FLAG_GELU | FLAG_BIAS);
    launch_conv(s, tA, p.w2, p.b2, p.g2, p.e2, tB, Cch, Cch, 1, 1);
    launch_gemm(s, tB, p.w3, p.b3, p.g3, p.e3, res, out, Cch, Cch, y_ctot, y_coff, 0,
                FLAG_LN | FLAG_BIAS | (res ? FLAG_RES : 0));
}

extern "C" void kernel_launch(void* const* d_in, const int* in_sizes, int n_in,
                              void* d_out, int out_size, void* d_ws, size_t ws_size,
                              hipStream_t stream)
{
    (void)in_sizes; (void)n_in; (void)out_size; (void)ws_size;

    const float* kernels = (const float*)d_in[0];
    const float* fm[4]   = {(const float*)d_in[1], (const float*)d_in[2],
                            (const float*)d_in[3], (const float*)d_in[4]};
    const int*   mask    = (const int*)d_in[5];

    DWP qp  = getdw(d_in, 6);
    DWP kp[4], vp[4];
    for (int m = 0; m < 4; ++m) { kp[m] = getdw(d_in, 18 + 12 * m); vp[m] = getdw(d_in, 66 + 12 * m); }
    DWP op  = getdw(d_in, 114);
    DWP fp  = getdw(d_in, 126);

    float* out = (float*)d_out;
    float* ws  = (float*)d_ws;

    const size_t CL = (size_t)Bn * Cch * Ls;   // 4,194,304 floats
    float* t1 = ws;                            // [B,64,L]
    float* t2 = ws + 1 * CL;                   // [B,64,L]
    float* qb = ws + 2 * CL;                   // [B,64,L]
    float* kb = ws + 3 * CL;                   // [B,256,L] (reused as ffn expand)
    float* vb = ws + 7 * CL;                   // [B,256,L] (reused as ffn conv out)
    float* ob = ws + 11 * CL;                  // [B,64,L]  (pre-ffn residual)
    float* at = ws + 12 * CL;                  // [B,64,256] attention

    // 1. q / k / v projections (DepthWiseConvBlocks)
    dwblock(stream, kernels, qp, t1, t2, qb, Cch, 0, nullptr);
    for (int m = 0; m < 4; ++m)
        dwblock(stream, fm[m], kp[m], t1, t2, kb, KCc, m * Cch, nullptr);
    for (int m = 0; m < 4; ++m)
        dwblock(stream, fm[m], vp[m], t1, t2, vb, KCc, m * Cch, nullptr);

    // 2. attention logits (split-K over L with f32 atomics), mask + softmax
    (void)hipMemsetAsync(at, 0, (size_t)Bn * Cch * KCc * sizeof(float), stream);
    {
        dim3 grid(Ls / LCHUNK, Bn);
        qk_kernel<<<grid, 256, 0, stream>>>(qb, kb, at);
    }
    softmax_mask_kernel<<<(Bn * Cch + 7) / 8, 256, 0, stream>>>(at, mask);

    // 3. x = attn @ v  (per-batch weight matrix, no bias / LN / act)
    launch_gemm(stream, vb, at, nullptr, nullptr, nullptr, nullptr, t1,
                KCc, Cch, Cch, 0, (long)Cch * KCc, 0);

    // 4. out_project DW block + identity residual -> ob
    dwblock(stream, t1, op, t2, t1, ob, Cch, 0, kernels);

    // 5. ffn GroupConvBlock (groups = 64, expand ratio 4), final GELU after residual
    launch_gemm(stream, ob, fp.w1, fp.b1, fp.g1, fp.e1, nullptr, kb,
                Cch, KCc, KCc, 0, 0, FLAG_LN | FLAG_GELU | FLAG_BIAS);
    launch_conv(stream, kb, fp.w2, fp.b2, fp.g2, fp.e2, vb, KCc, KCc, 4, 4);
    launch_gemm(stream, vb, fp.w3, fp.b3, fp.g3, fp.e3, ob, out,
                KCc, Cch, Cch, 0, 0, FLAG_LN | FLAG_GELU | FLAG_BIAS | FLAG_RES);

    // 6. feature maps pass through unchanged (tuple output)
    for (int m = 0; m < 4; ++m)
        (void)hipMemcpyAsync(out + CL * (1 + m), fm[m], CL * sizeof(float),
                             hipMemcpyDeviceToDevice, stream);
}
